// Part_Graph_60464549593222
// MI455X (gfx1250) — matrix operand order repaired and checked by reading, not compiled, not run on hardware
//
#include <hip/hip_runtime.h>
#include <math.h>

typedef _Float16 f16;
typedef __attribute__((ext_vector_type(2))) _Float16 v2h;
typedef __attribute__((ext_vector_type(8))) _Float16 v8h;
typedef __attribute__((ext_vector_type(16))) _Float16 v16h;
typedef __attribute__((ext_vector_type(8))) float v8f;
typedef __attribute__((ext_vector_type(4))) unsigned int v4u;
typedef __attribute__((ext_vector_type(8))) int v8i;
typedef __attribute__((ext_vector_type(4))) int v4i;

typedef const __attribute__((address_space(1))) float* gcf_p;
typedef __attribute__((address_space(1))) float* gf_p;
typedef __attribute__((address_space(1))) f16* gh_p;
typedef const __attribute__((address_space(1))) v4u* gc4_p;

#define N_IMG 16
#define HD 10
#define HW 16384
#define EPS_F 1e-5f
#define PSTR 152   // pixT per-pixel stride (halves); 304B, 16B-aligned, conflict-free
#define SCR 112    // scratch segment start (halves)

// ---- f16 weight arena offsets (halves) ----
#define WH_Wa1_u 0      // 16 x 72  (K=50 pad 64, stride 72)
#define WH_Wa2_u 1152   // 16 x 40  (K=10 pad 32, stride 40)
#define WH_Wd_u  1792
#define WH_Wa1_l 2432
#define WH_Wa2_l 3072
#define WH_Wd_l  3712
#define WH_Wg    4352   // 6 x (32 x 40)
#define WH_Wc    12032  // 6 x (16 x 40)
#define WH_TOT   15872  // 31744 B = 1984 uint4
#define WS_PF_U4 2048   // params at byte 32768 in d_ws

// ---- f32 param offsets (floats) ----
#define PF_s1_u 0
#define PF_be1_u 10
#define PF_sd_u 20
#define PF_bed_u 30
#define PF_s1_l 40
#define PF_be1_l 50
#define PF_sd_l 60
#define PF_bed_l 70
#define PF_ba2_u 80
#define PF_ba2_l 85
#define PF_bg 88
#define PF_bc 208
#define PF_TOT 268      // 1072 B = 67 uint4

__device__ __forceinline__ v8f wmmaH(v16h a, v16h b, v8f c) {
  return __builtin_amdgcn_wmma_f32_16x16x32_f16(false, a, false, b, (short)0, c, false, false);
}
__device__ __forceinline__ v16h cat16(v8h lo, v8h hi) {
  return __builtin_shufflevector(lo, hi, 0, 1, 2, 3, 4, 5, 6, 7, 8, 9, 10, 11, 12, 13, 14, 15);
}
__device__ __forceinline__ v16h ldA(const f16* W, int stride, int m, int jj, int kb8) {
  const f16* p = W + m * stride + jj * 32 + 2 * kb8;
  return cat16(*(const v8h*)p, *(const v8h*)(p + 8));
}
__device__ __forceinline__ v16h ldB(const f16* prow, int seg, int kb16) {
  const f16* p = prow + seg + kb16;
  return cat16(*(const v8h*)p, *(const v8h*)(p + 8));
}
__device__ __forceinline__ float sigf(float x) { return 1.f / (1.f + __expf(-x)); }
__device__ __forceinline__ float tanhfast(float x) { return 1.f - 2.f / (__expf(2.f * x) + 1.f); }
__device__ __forceinline__ v2h pk2(float a, float b) {
  __fp16 __attribute__((ext_vector_type(2))) t = __builtin_amdgcn_cvt_pkrtz(a, b);
  v2h r;
  __builtin_memcpy(&r, &t, sizeof(r));
  return r;
}

struct Ctx {
  int q, m, hi, kb8, kb16;
  int n, hw;
  gf_p out;
  f16* prow;
  float (*pixf)[16];
  float (*uT)[16];
  const f16* wH;
  const float* pf;
};

// ---- TDM: 3-D tile (tile0 contiguous, tile1 @stride0, tile2 @stride1), f32 elems ----
__device__ __forceinline__ void tdm_load_3d(unsigned lds_addr, unsigned long long gaddr,
                                            int tile0, int tile1, int tile2,
                                            long long stride0, long long stride1) {
#if __has_builtin(__builtin_amdgcn_tensor_load_to_lds)
  const unsigned td = 0x40000000u;
  v4u g0;
  g0[0] = 1u;
  g0[1] = lds_addr;
  g0[2] = (unsigned)(gaddr & 0xffffffffu);
  g0[3] = (unsigned)((gaddr >> 32) & 0x01ffffffu) | (2u << 30);
  v8i g1;
  g1[0] = (int)(2u << 16);
  g1[1] = (int)((td & 0xffffu) << 16);
  g1[2] = (int)((td >> 16) | ((td & 0xffffu) << 16));
  g1[3] = (int)((td >> 16) | ((unsigned)tile0 << 16));
  g1[4] = (int)((unsigned)tile1 | ((unsigned)tile2 << 16));
  g1[5] = (int)((unsigned long long)stride0 & 0xffffffffu);
  g1[6] = (int)((((unsigned long long)stride0 >> 32) & 0xffffu) |
                (((unsigned long long)stride1 & 0xffffu) << 16));
  g1[7] = (int)(((unsigned long long)stride1 >> 16) & 0xffffffffu);
  v4i g2;
  g2[0] = 0x40000000;
  g2[1] = 0;
  g2[2] = 0;
  g2[3] = 0;
  v4i g3 = {0, 0, 0, 0};
  v8i g4 = {0, 0, 0, 0, 0, 0, 0, 0};
  __builtin_amdgcn_tensor_load_to_lds(g0, g1, g2, g3, g4, 0);
#endif
}

__device__ __forceinline__ void tdm_wait() {
#if __has_builtin(__builtin_amdgcn_tensor_load_to_lds)
#if __has_builtin(__builtin_amdgcn_s_wait_tensorcnt)
  __builtin_amdgcn_s_wait_tensorcnt((short)0);
#else
  asm volatile("s_wait_tensorcnt 0x0" ::: "memory");
#endif
  asm volatile("" ::: "memory");
#endif
}

// =====================  setup kernel: build f16 arena + params in d_ws  =====================
__device__ __forceinline__ void cpw16g(gh_p dst, const float* src, int O, int Cc, int stride,
                                       int rows, int tid) {
  gcf_p s = (gcf_p)(unsigned long long)(const void*)src;
  for (int t = tid; t < rows * stride; t += 256) {
    int mm = t / stride, k = t % stride;
    float v = (mm < O && k < Cc) ? s[mm * Cc + k] : 0.f;
    dst[t] = (f16)v;
  }
}

__global__ __launch_bounds__(256) void setup_weights(
    const float* Wa1_u, const float* Wa2_u, const float* Wd_u,
    const float* Wa1_l, const float* Wa2_l, const float* Wd_l,
    const float* Wg, const float* Wc,
    const float* ba1_u, const float* g1_u, const float* b1_u, const float* m1_u,
    const float* v1_u, const float* ba2_u,
    const float* gd_u, const float* bd_u, const float* md_u, const float* vd_u,
    const float* ba1_l, const float* g1_l, const float* b1_l, const float* m1_l,
    const float* v1_l, const float* ba2_l,
    const float* gd_l, const float* bd_l, const float* md_l, const float* vd_l,
    const float* bg, const float* bc, void* ws) {
  int tid = threadIdx.x;
  gh_p wsH = (gh_p)(unsigned long long)ws;
  gf_p wsP = (gf_p)((unsigned long long)ws + 32768ull);

  cpw16g(wsH + WH_Wa1_u, Wa1_u, 10, 50, 72, 16, tid);
  cpw16g(wsH + WH_Wa2_u, Wa2_u, 5, 10, 40, 16, tid);
  cpw16g(wsH + WH_Wd_u, Wd_u, 10, 10, 40, 16, tid);
  cpw16g(wsH + WH_Wa1_l, Wa1_l, 10, 30, 40, 16, tid);
  cpw16g(wsH + WH_Wa2_l, Wa2_l, 3, 10, 40, 16, tid);
  cpw16g(wsH + WH_Wd_l, Wd_l, 10, 10, 40, 16, tid);
#pragma unroll
  for (int i = 0; i < 6; ++i) {
    cpw16g(wsH + WH_Wg + i * 1280, Wg + i * 400, 20, 20, 40, 32, tid);
    cpw16g(wsH + WH_Wc + i * 640, Wc + i * 200, 10, 20, 40, 16, tid);
  }
  if (tid < 10) {
    gcf_p G = (gcf_p)(unsigned long long)(const void*)g1_u;
    gcf_p V = (gcf_p)(unsigned long long)(const void*)v1_u;
    gcf_p A = (gcf_p)(unsigned long long)(const void*)ba1_u;
    gcf_p B = (gcf_p)(unsigned long long)(const void*)b1_u;
    gcf_p M = (gcf_p)(unsigned long long)(const void*)m1_u;
    float s = G[tid] * rsqrtf(V[tid] + EPS_F);
    wsP[PF_s1_u + tid] = s;
    wsP[PF_be1_u + tid] = A[tid] * s + B[tid] - M[tid] * s;
    gcf_p Gd = (gcf_p)(unsigned long long)(const void*)gd_u;
    gcf_p Vd = (gcf_p)(unsigned long long)(const void*)vd_u;
    gcf_p Bd = (gcf_p)(unsigned long long)(const void*)bd_u;
    gcf_p Md = (gcf_p)(unsigned long long)(const void*)md_u;
    float t2 = Gd[tid] * rsqrtf(Vd[tid] + EPS_F);
    wsP[PF_sd_u + tid] = t2;
    wsP[PF_bed_u + tid] = Bd[tid] - Md[tid] * t2;
    gcf_p G2 = (gcf_p)(unsigned long long)(const void*)g1_l;
    gcf_p V2 = (gcf_p)(unsigned long long)(const void*)v1_l;
    gcf_p A2 = (gcf_p)(unsigned long long)(const void*)ba1_l;
    gcf_p B2 = (gcf_p)(unsigned long long)(const void*)b1_l;
    gcf_p M2 = (gcf_p)(unsigned long long)(const void*)m1_l;
    float s2 = G2[tid] * rsqrtf(V2[tid] + EPS_F);
    wsP[PF_s1_l + tid] = s2;
    wsP[PF_be1_l + tid] = A2[tid] * s2 + B2[tid] - M2[tid] * s2;
    gcf_p Gd2 = (gcf_p)(unsigned long long)(const void*)gd_l;
    gcf_p Vd2 = (gcf_p)(unsigned long long)(const void*)vd_l;
    gcf_p Bd2 = (gcf_p)(unsigned long long)(const void*)bd_l;
    gcf_p Md2 = (gcf_p)(unsigned long long)(const void*)md_l;
    float t3 = Gd2[tid] * rsqrtf(Vd2[tid] + EPS_F);
    wsP[PF_sd_l + tid] = t3;
    wsP[PF_bed_l + tid] = Bd2[tid] - Md2[tid] * t3;
  }
  if (tid < 5) wsP[PF_ba2_u + tid] = ((gcf_p)(unsigned long long)(const void*)ba2_u)[tid];
  if (tid < 3) wsP[PF_ba2_l + tid] = ((gcf_p)(unsigned long long)(const void*)ba2_l)[tid];
  if (tid < 120) wsP[PF_bg + tid] = ((gcf_p)(unsigned long long)(const void*)bg)[tid];
  if (tid < 60) wsP[PF_bc + tid] = ((gcf_p)(unsigned long long)(const void*)bc)[tid];
}

// =====================  main fused kernel  =====================
template <int P, int GRU0, int SEG_F, int NJ, int A1STR>
__device__ __forceinline__ void decomp_gru(const Ctx& C,
    const f16* wa1, const f16* wa2, const f16* wd,
    const float* s1, const float* be1, const float* ba2,
    const float* sd, const float* bed, gf_p maps_out) {
  v8f z = {};

  // ---- t = relu(bn(Wa1 @ feats)) ----
  v8f acc = z;
#pragma unroll
  for (int jj = 0; jj < NJ; ++jj)
    acc = wmmaH(ldA(wa1, A1STR, C.m, jj, C.kb8), ldB(C.prow, SEG_F + 32 * jj, C.kb16), acc);
  if (!C.hi) {
#pragma unroll
    for (int e = 0; e < 4; ++e) {
      float x = fmaxf(acc[2 * e] * s1[2 * e] + be1[2 * e], 0.f);
      float y = fmaxf(acc[2 * e + 1] * s1[2 * e + 1] + be1[2 * e + 1], 0.f);
      *(v2h*)(C.prow + SCR + 2 * e) = pk2(x, y);
    }
    *(v2h*)(C.prow + SCR + 10) = pk2(0.f, 0.f);
    *(v2h*)(C.prow + SCR + 12) = pk2(0.f, 0.f);
    *(v2h*)(C.prow + SCR + 14) = pk2(0.f, 0.f);
  } else {
    float x = fmaxf(acc[0] * s1[8] + be1[8], 0.f);
    float y = fmaxf(acc[1] * s1[9] + be1[9], 0.f);
    *(v2h*)(C.prow + SCR + 8) = pk2(x, y);
    *(v2h*)(C.prow + SCR + 16) = pk2(0.f, 0.f);
    *(v2h*)(C.prow + SCR + 18) = pk2(0.f, 0.f);
  }

  // ---- maps = Wa2 @ t + ba2 ----
  v8f mp = wmmaH(ldA(wa2, 40, C.m, 0, C.kb8), ldB(C.prow, SCR, C.kb16), z);

  // ---- softmax over P+1 rows (all in lanes 0-15) ----
  float att[P + 1] = {};
  if (!C.hi) {
    float mv[P + 1];
    float mx = -1e30f;
#pragma unroll
    for (int k = 0; k <= P; ++k) { mv[k] = mp[k] + ba2[k]; mx = fmaxf(mx, mv[k]); }
    float sum = 0.f;
#pragma unroll
    for (int k = 0; k <= P; ++k) { att[k] = __expf(mv[k] - mx); sum += att[k]; }
    float inv = 1.f / sum;
    gf_p mo = maps_out + (size_t)(C.n * (P + 1)) * HW + C.hw;
#pragma unroll
    for (int k = 0; k <= P; ++k) {
      att[k] *= inv;
      mo[k * HW] = mv[k];
    }
  }

  // ---- per part: gated decode + GRU cell ----
#pragma unroll
  for (int p = 0; p < P; ++p) {
    const int i = GRU0 + p;
    const int hoff = (i == 0) ? 96 : (i <= 4 ? 10 * i : 74);
    float ab = __shfl(att[1 + p], C.q, 32);

    // stage gated = xf*att + part -> SCR[0..9], zeros [10..19]
    if (!C.hi) {
#pragma unroll
      for (int e = 0; e < 3; ++e) {
        v2h xf = *(const v2h*)(C.prow + SEG_F + 2 * e);
        v2h pt = *(const v2h*)(C.prow + SEG_F + 10 + 10 * p + 2 * e);
        *(v2h*)(C.prow + SCR + 2 * e) =
            pk2((float)xf.x * ab + (float)pt.x, (float)xf.y * ab + (float)pt.y);
      }
      *(v2h*)(C.prow + SCR + 10) = pk2(0.f, 0.f);
      *(v2h*)(C.prow + SCR + 12) = pk2(0.f, 0.f);
      *(v2h*)(C.prow + SCR + 14) = pk2(0.f, 0.f);
    } else {
#pragma unroll
      for (int e = 3; e < 5; ++e) {
        v2h xf = *(const v2h*)(C.prow + SEG_F + 2 * e);
        v2h pt = *(const v2h*)(C.prow + SEG_F + 10 + 10 * p + 2 * e);
        *(v2h*)(C.prow + SCR + 2 * e) =
            pk2((float)xf.x * ab + (float)pt.x, (float)xf.y * ab + (float)pt.y);
      }
      *(v2h*)(C.prow + SCR + 16) = pk2(0.f, 0.f);
      *(v2h*)(C.prow + SCR + 18) = pk2(0.f, 0.f);
    }

    // msg -> SCR[0..9]; h -> SCR[10..19]
    v8f od = wmmaH(ldA(wd, 40, C.m, 0, C.kb8), ldB(C.prow, SCR, C.kb16), z);
    if (!C.hi) {
#pragma unroll
      for (int e = 0; e < 4; ++e) {
        float x = fmaxf(od[2 * e] * sd[2 * e] + bed[2 * e], 0.f);
        float y = fmaxf(od[2 * e + 1] * sd[2 * e + 1] + bed[2 * e + 1], 0.f);
        *(v2h*)(C.prow + SCR + 2 * e) = pk2(x, y);
      }
#pragma unroll
      for (int e = 0; e < 3; ++e)
        *(v2h*)(C.prow + SCR + 10 + 2 * e) = *(const v2h*)(C.prow + hoff + 2 * e);
    } else {
      float x = fmaxf(od[0] * sd[8] + bed[8], 0.f);
      float y = fmaxf(od[1] * sd[9] + bed[9], 0.f);
      *(v2h*)(C.prow + SCR + 8) = pk2(x, y);
#pragma unroll
      for (int e = 0; e < 2; ++e)
        *(v2h*)(C.prow + SCR + 16 + 2 * e) = *(const v2h*)(C.prow + hoff + 6 + 2 * e);
    }

    // gates
    const f16* wg = C.wH + WH_Wg + i * 1280;
    const float* bgp = C.pf + PF_bg + i * 20;
    v16h bB = ldB(C.prow, SCR, C.kb16);
    v8f gA = wmmaH(ldA(wg, 40, C.m, 0, C.kb8), bB, z);
    v8f gB = wmmaH(ldA(wg, 40, C.m + 16, 0, C.kb8), bB, z);
    if (!C.hi) {
#pragma unroll
      for (int e = 0; e < 4; ++e) {
        float r0 = sigf(gA[2 * e] + bgp[2 * e]);
        float r1 = sigf(gA[2 * e + 1] + bgp[2 * e + 1]);
        v2h hh = *(const v2h*)(C.prow + hoff + 2 * e);
        *(v2h*)(C.prow + SCR + 10 + 2 * e) = pk2(r0 * (float)hh.x, r1 * (float)hh.y);
      }
#pragma unroll
      for (int r = 0; r < 4; ++r) C.uT[6 + r][C.q] = sigf(gB[r] + bgp[16 + r]);
    } else {
      float r8 = sigf(gA[0] + bgp[8]);
      float r9 = sigf(gA[1] + bgp[9]);
      v2h hh = *(const v2h*)(C.prow + hoff + 8);
      *(v2h*)(C.prow + SCR + 18) = pk2(r8 * (float)hh.x, r9 * (float)hh.y);
#pragma unroll
      for (int r = 2; r < 8; ++r) C.uT[r - 2][C.q] = sigf(gA[r] + bgp[8 + r]);
    }

    // cand + update
    const f16* wc = C.wH + WH_Wc + i * 640;
    const float* bcp = C.pf + PF_bc + i * 10;
    v8f cg = wmmaH(ldA(wc, 40, C.m, 0, C.kb8), ldB(C.prow, SCR, C.kb16), z);
    gf_p ob = C.out + ((size_t)((1 + i) * N_IMG + C.n) * HD + (C.hi ? 8 : 0)) * HW + C.hw;
    if (!C.hi) {
#pragma unroll
      for (int r = 0; r < 8; ++r) {
        float cv = tanhfast(cg[r] + bcp[r]);
        float u = C.uT[r][C.q];
        float h = C.pixf[i * 10 + r][C.q];
        ob[r * HW] = fmaf(u, cv - h, h);
      }
    } else {
#pragma unroll
      for (int r = 0; r < 2; ++r) {
        float cv = tanhfast(cg[r] + bcp[8 + r]);
        float u = C.uT[8 + r][C.q];
        float h = C.pixf[i * 10 + 8 + r][C.q];
        ob[r * HW] = fmaf(u, cv - h, h);
      }
    }
  }
}

__global__ __launch_bounds__(256) void fused_part_graph(
    const float* xh, const float* xp, const void* ws, float* out) {
  __shared__ __align__(16) f16 wH[WH_TOT];
  __shared__ __align__(16) float pf[PF_TOT];
  __shared__ float pixf[8][90][16];
  __shared__ __align__(16) f16 pixT[8][16 * PSTR];
  __shared__ float uT[8][10][16];

  int tid = threadIdx.x;
  int wslot = tid >> 5;
  int lane = tid & 31;

  Ctx C;
  C.q = lane & 15;
  C.m = lane & 15;
  C.hi = (lane >= 16) ? 1 : 0;
  C.kb8 = C.hi ? 8 : 0;
  C.kb16 = C.hi ? 16 : 0;
  int wave = blockIdx.x * 8 + wslot;
  int g = wave * 16 + C.q;
  C.n = g >> 14;
  C.hw = g & (HW - 1);
  C.out = (gf_p)(unsigned long long)(void*)out;
  C.prow = &pixT[wslot][C.q * PSTR];
  C.pixf = pixf[wslot];
  C.uT = uT[wslot];
  C.wH = wH;
  C.pf = pf;

  int hw0 = C.hw - C.q;

  // ---- per-wave TDM tile loads: xp (16px x 10ch x 7 parts), xh (x 2) ----
#if __has_builtin(__builtin_amdgcn_tensor_load_to_lds)
  {
    unsigned lds_xp = (unsigned)(unsigned long long)(void*)&pixf[wslot][0][0];
    unsigned lds_xh = (unsigned)(unsigned long long)(void*)&pixf[wslot][70][0];
    unsigned long long base = ((unsigned long long)(C.n * HD) * HW + (unsigned)hw0) * 4ull;
    unsigned long long gxp = (unsigned long long)(const void*)xp + base;
    unsigned long long gxh = (unsigned long long)(const void*)xh + base;
    tdm_load_3d(lds_xp, gxp, 16, 10, 7, (long long)HW, 160LL * HW);
    tdm_load_3d(lds_xh, gxh, 16, 10, 2, (long long)HW, 160LL * HW);
  }
#else
  {
    gcf_p xpg = (gcf_p)(unsigned long long)(const void*)xp;
    gcf_p xhg = (gcf_p)(unsigned long long)(const void*)xh;
    size_t base = (size_t)(C.n * HD) * HW + (unsigned)hw0;
    for (int t = lane; t < 70 * 16; t += 32) {
      int row = t >> 4, px = t & 15;
      pixf[wslot][row][px] = xpg[base + (size_t)((row / 10) * 160 + row % 10) * HW + px];
    }
    for (int t = lane; t < 20 * 16; t += 32) {
      int row = t >> 4, px = t & 15;
      pixf[wslot][70 + row][px] = xhg[base + (size_t)((row / 10) * 160 + row % 10) * HW + px];
    }
  }
#endif

  // ---- copy prebuilt f16 arena + params from d_ws into LDS (vectorized) ----
  {
    gc4_p wsrc = (gc4_p)(unsigned long long)ws;
    v4u* dstW = (v4u*)wH;
    for (int t = tid; t < WH_TOT / 8; t += 256) dstW[t] = wsrc[t];
    v4u* dstP = (v4u*)pf;
    for (int t = tid; t < PF_TOT / 4; t += 256) dstP[t] = wsrc[WS_PF_U4 + t];
  }
  __syncthreads();
  tdm_wait();

  // ---- transpose f32 tile -> pixel-major f16 tile, zero-pad K segments ----
#pragma unroll
  for (int it = 0; it < 38; ++it) {
    int d = (C.hi ? 2 : 0) + 4 * it;
    int row = 0;
    bool valid = true;
    if (d < 50)        row = (d < 10) ? 70 + d : d;
    else if (d < 64)   valid = false;
    else if (d < 74)   row = 80 + (d - 64);
    else if (d < 94)   row = 50 + (d - 74);
    else if (d < 96)   valid = false;
    else if (d < 106)  row = d - 96;
    else               valid = false;
    float a = 0.f, b2 = 0.f;
    if (valid) { a = C.pixf[row][C.q]; b2 = C.pixf[row + 1][C.q]; }
    *(v2h*)(C.prow + d) = pk2(a, b2);
  }

  // ---- xp_new[0] = xp[0] ----
  {
    gf_p ob = C.out + ((size_t)(0 * N_IMG + C.n) * HD + (C.hi ? 5 : 0)) * HW + C.hw;
#pragma unroll
    for (int c0 = 0; c0 < 5; ++c0) ob[c0 * HW] = C.pixf[c0 + (C.hi ? 5 : 0)][C.q];
  }

  decomp_gru<4, 0, 0, 2, 72>(C, wH + WH_Wa1_u, wH + WH_Wa2_u, wH + WH_Wd_u,
      pf + PF_s1_u, pf + PF_be1_u, pf + PF_ba2_u, pf + PF_sd_u, pf + PF_bed_u,
      C.out + (size_t)18350080);
  decomp_gru<2, 4, 64, 1, 40>(C, wH + WH_Wa1_l, wH + WH_Wa2_l, wH + WH_Wd_l,
      pf + PF_s1_l, pf + PF_be1_l, pf + PF_ba2_l, pf + PF_sd_l, pf + PF_bed_l,
      C.out + (size_t)19660800);
}

extern "C" void kernel_launch(void* const* d_in, const int* in_sizes, int n_in,
                              void* d_out, int out_size, void* d_ws, size_t ws_size,
                              hipStream_t stream) {
  (void)in_sizes; (void)n_in; (void)out_size; (void)ws_size;
  const float* A[32];
  for (int i = 0; i < 32; ++i) A[i] = (const float*)d_in[i];
  setup_weights<<<dim3(1), dim3(256), 0, stream>>>(
      A[2], A[8], A[10], A[15], A[21], A[23], A[28], A[30],
      A[3], A[4], A[5], A[6], A[7], A[9], A[11], A[12], A[13], A[14],
      A[16], A[17], A[18], A[19], A[20], A[22], A[24], A[25], A[26], A[27],
      A[29], A[31], d_ws);
  fused_part_graph<<<dim3(2048), dim3(256), 0, stream>>>(A[0], A[1], d_ws, (float*)d_out);
}
